// ShortCutQ_30305289240887
// MI455X (gfx1250) — 8-bit matrix operand order repaired and checked by reading, compiled, not run on hardware
//
#include <hip/hip_runtime.h>
#include <stdint.h>

typedef int   v16i __attribute__((ext_vector_type(16)));
typedef float v8f  __attribute__((ext_vector_type(8)));

#define B_N      16
#define C_N      512
#define O_N      1024
#define H_IN     28
#define H_OUT    14
#define SP_N     196
#define NTOT     3136
#define NT_TILES 196
#define MT_TILES 16
#define S_N      128

#define INV3D    (63.0f / 1280.0f)
#define DELTA    (1280.0f / 189.0f)

#define FP8_ONE  0x38u

static __device__ __forceinline__ float bn_relu(float xv, float g, float b, float m, float v) {
    float inv = rsqrtf(v + 1e-5f);
    float r = (xv - m) * (inv * g) + b;
    return fmaxf(r, 0.0f);
}

__global__ void k_amax_x(const float* __restrict__ x,
                         const float* __restrict__ gam, const float* __restrict__ bet,
                         const float* __restrict__ mea, const float* __restrict__ var,
                         unsigned int* __restrict__ outbits) {
    __shared__ float sm[256];
    float mx = 0.0f;
    const long total = (long)B_N * C_N * H_IN * H_IN;
    for (long i = (long)blockIdx.x * blockDim.x + threadIdx.x; i < total;
         i += (long)gridDim.x * blockDim.x) {
        int c = (int)((i / (H_IN * H_IN)) % C_N);
        mx = fmaxf(mx, bn_relu(x[i], gam[c], bet[c], mea[c], var[c]));
    }
    sm[threadIdx.x] = mx;
    __syncthreads();
    for (int s = 128; s > 0; s >>= 1) {
        if ((int)threadIdx.x < s) sm[threadIdx.x] = fmaxf(sm[threadIdx.x], sm[threadIdx.x + s]);
        __syncthreads();
    }
    if (threadIdx.x == 0) atomicMax(outbits, __float_as_uint(sm[0]));
}

__global__ void k_amax_w(const float* __restrict__ w, unsigned int* __restrict__ outbits) {
    __shared__ float sm[256];
    float mx = 0.0f;
    const int total = O_N * C_N;
    for (int i = blockIdx.x * blockDim.x + threadIdx.x; i < total;
         i += gridDim.x * blockDim.x) {
        mx = fmaxf(mx, fabsf(w[i]));
    }
    sm[threadIdx.x] = mx;
    __syncthreads();
    for (int s = 128; s > 0; s >>= 1) {
        if ((int)threadIdx.x < s) sm[threadIdx.x] = fmaxf(sm[threadIdx.x], sm[threadIdx.x + s]);
        __syncthreads();
    }
    if (threadIdx.x == 0) atomicMax(outbits, __float_as_uint(sm[0]));
}

__global__ void k_quant_x(const float* __restrict__ x,
                          const float* __restrict__ gam, const float* __restrict__ bet,
                          const float* __restrict__ mea, const float* __restrict__ var,
                          const unsigned int* __restrict__ amaxbits,
                          uint8_t* __restrict__ xq, uint8_t* __restrict__ xb) {
    int tid = blockIdx.x * blockDim.x + threadIdx.x;
    if (tid >= NTOT * C_N) return;
    float amax = fmaxf(__uint_as_float(*amaxbits), 1e-8f);
    int c  = tid & (C_N - 1);
    int n  = tid >> 9;
    int b  = n / SP_N;
    int sp = n - b * SP_N;
    int ho = sp / H_OUT;
    int wo = sp - ho * H_OUT;
    long xi = (((long)b * C_N + c) * H_IN + 2 * ho) * H_IN + 2 * wo;
    float v = bn_relu(x[xi], gam[c], bet[c], mea[c], var[c]);
    int q = (int)rintf(fminf(fmaxf(v / amax, 0.0f), 1.0f) * 255.0f);
    xq[(long)n * C_N + c] = (uint8_t)q;
    int g = c >> 7;
    int s = c & 127;
    #pragma unroll
    for (int z = 0; z < 8; ++z) {
        uint8_t code = (uint8_t)(((q >> z) & 1) * FP8_ONE);
        xb[((long)(g * 8 + z) * NTOT + n) * S_N + s] = code;
    }
}

__global__ void k_xstats(const uint8_t* __restrict__ xq,
                         float* __restrict__ dummyacc) {
    int n = blockIdx.x * blockDim.x + threadIdx.x;
    if (n >= NTOT) return;
    const uint32_t* p = (const uint32_t*)(xq + (long)n * C_N);
    float dacc = 0.0f;
    for (int g = 0; g < 4; ++g) {
        int sums[8] = {0, 0, 0, 0, 0, 0, 0, 0};
        #pragma unroll
        for (int wd = 0; wd < 32; ++wd) {
            uint32_t v = p[g * 32 + wd];
            #pragma unroll
            for (int z = 0; z < 8; ++z) sums[z] += __popc((v >> z) & 0x01010101u);
        }
        #pragma unroll
        for (int z = 0; z < 8; ++z) {
            float xz = (float)sums[z];
            float d = 21.0f * rintf(xz * INV3D) + 64.0f * rintf(7.0f * xz * INV3D);
            dacc += (float)(1 << z) * d;
        }
    }
    dummyacc[n] = dacc;
}

__global__ void k_quant_w(const float* __restrict__ w,
                          const unsigned int* __restrict__ wmaxbits,
                          uint8_t* __restrict__ wcell) {
    int tid = blockIdx.x * blockDim.x + threadIdx.x;
    if (tid >= O_N * C_N) return;
    float wmax = fmaxf(__uint_as_float(*wmaxbits), 1e-8f);
    float wi = rintf(fminf(fmaxf(w[tid] / wmax, -1.0f), 1.0f) * 127.0f);
    int wu = (int)wi + 128;
    #pragma unroll
    for (int k = 0; k < 4; ++k) {
        int cell = (wu >> (2 * k)) & 3;
        uint8_t code = (cell < 2) ? (cell ? 0x48 : 0x38) : ((cell == 2) ? 0x4E : 0x52);
        wcell[(long)k * O_N * C_N + tid] = code;
    }
}

__global__ __launch_bounds__(256) void k_cim_gemm(
    const uint8_t* __restrict__ xb, const uint8_t* __restrict__ wcell,
    const float* __restrict__ dummyacc,
    const unsigned int* __restrict__ amaxbits, const unsigned int* __restrict__ wmaxbits,
    const float* __restrict__ y, float* __restrict__ out) {

    const int lane   = threadIdx.x & 31;
    const int waveId = (blockIdx.x * blockDim.x + threadIdx.x) >> 5;
    const int mt  = waveId / NT_TILES;
    const int nt  = waveId - mt * NT_TILES;
    const int hl  = lane >> 4;
    const int l15 = lane & 15;
    const int kh16  = hl << 4;
    const int ah8   = hl << 4;
    const int rhalf = hl << 3;

    float facc[4][8];
    #pragma unroll
    for (int m = 0; m < 4; ++m)
        #pragma unroll
        for (int i = 0; i < 8; ++i) facc[m][i] = 0.0f;

    const int ncol = nt * 16 + l15;

    for (int g = 0; g < 4; ++g) {
        #pragma unroll
        for (int m = 0; m < 4; ++m) {
            const int row = mt * 64 + m * 16 + l15;
            const uint8_t* ab = wcell + (long)row * C_N + g * S_N + ah8;
            #pragma unroll
            for (int k = 0; k < 4; ++k) {
                const uint8_t* ak = ab + (long)k * O_N * C_N;
                uint4 a0 = *(const uint4*)(ak + 0);
                uint4 a1 = *(const uint4*)(ak + 32);
                uint4 a2 = *(const uint4*)(ak + 64);
                uint4 a3 = *(const uint4*)(ak + 96);
                v16i a;
                a[0]  = (int)a0.x; a[1]  = (int)a0.y; a[2]  = (int)a0.z; a[3]  = (int)a0.w;
                a[4]  = (int)a1.x; a[5]  = (int)a1.y; a[6]  = (int)a1.z; a[7]  = (int)a1.w;
                a[8]  = (int)a2.x; a[9]  = (int)a2.y; a[10] = (int)a2.z; a[11] = (int)a2.w;
                a[12] = (int)a3.x; a[13] = (int)a3.y; a[14] = (int)a3.z; a[15] = (int)a3.w;

                for (int z = 0; z < 8; ++z) {
                    const uint8_t* bp =
                        xb + ((long)(g * 8 + z) * NTOT + ncol) * S_N + kh16;
                    uint4 q0 = *(const uint4*)(bp + 0);
                    uint4 q1 = *(const uint4*)(bp + 32);
                    uint4 q2 = *(const uint4*)(bp + 64);
                    uint4 q3 = *(const uint4*)(bp + 96);
                    v16i bfrag;
                    bfrag[0]  = (int)q0.x; bfrag[1]  = (int)q0.y;
                    bfrag[2]  = (int)q0.z; bfrag[3]  = (int)q0.w;
                    bfrag[4]  = (int)q1.x; bfrag[5]  = (int)q1.y;
                    bfrag[6]  = (int)q1.z; bfrag[7]  = (int)q1.w;
                    bfrag[8]  = (int)q2.x; bfrag[9]  = (int)q2.y;
                    bfrag[10] = (int)q2.z; bfrag[11] = (int)q2.w;
                    bfrag[12] = (int)q3.x; bfrag[13] = (int)q3.y;
                    bfrag[14] = (int)q3.z; bfrag[15] = (int)q3.w;

                    v8f d = {};
                    d = __builtin_amdgcn_wmma_f32_16x16x128_fp8_fp8(a, bfrag, (short)0, d,
                                                                    false, false);
                    float wzk = (float)(1 << (z + 2 * k));
                    #pragma unroll
                    for (int i = 0; i < 8; ++i) {
                        facc[m][i] += wzk * rintf(d[i] * INV3D);
                    }
                }
            }
        }
    }

    float amax  = fmaxf(__uint_as_float(*amaxbits), 1e-8f);
    float wmax  = fmaxf(__uint_as_float(*wmaxbits), 1e-8f);
    float scale = (wmax / 127.0f) * (amax / 255.0f);
    float dsum  = dummyacc[ncol];
    int b  = ncol / SP_N;
    int sp = ncol - b * SP_N;
    #pragma unroll
    for (int m = 0; m < 4; ++m) {
        #pragma unroll
        for (int i = 0; i < 8; ++i) {
            int o = mt * 64 + m * 16 + rhalf + i;
            long oi = ((long)b * O_N + o) * SP_N + sp;
            out[oi] = (DELTA * (facc[m][i] - dsum)) * scale + y[oi];
        }
    }
}

extern "C" void kernel_launch(void* const* d_in, const int* in_sizes, int n_in,
                              void* d_out, int out_size, void* d_ws, size_t ws_size,
                              hipStream_t stream) {
    const float* x   = (const float*)d_in[0];
    const float* y   = (const float*)d_in[1];
    const float* gam = (const float*)d_in[2];
    const float* bet = (const float*)d_in[3];
    const float* mea = (const float*)d_in[4];
    const float* var = (const float*)d_in[5];
    const float* w   = (const float*)d_in[6];
    float* out = (float*)d_out;

    char* ws = (char*)d_ws;
    unsigned int* amaxb = (unsigned int*)(ws + 0);
    unsigned int* wmaxb = (unsigned int*)(ws + 4);
    uint8_t* xb    = (uint8_t*)(ws + 128);
    uint8_t* xq    = (uint8_t*)(ws + 128 + 12845056);
    uint8_t* wcell = (uint8_t*)(ws + 128 + 12845056 + 1605632);
    float*   dacc  = (float*)  (ws + 128 + 12845056 + 1605632 + 2097152);

    hipMemsetAsync(d_ws, 0, 8, stream);

    k_amax_x<<<1024, 256, 0, stream>>>(x, gam, bet, mea, var, amaxb);
    k_amax_w<<<256, 256, 0, stream>>>(w, wmaxb);
    k_quant_x<<<(NTOT * C_N + 255) / 256, 256, 0, stream>>>(x, gam, bet, mea, var, amaxb,
                                                            xq, xb);
    k_xstats<<<(NTOT + 255) / 256, 256, 0, stream>>>(xq, dacc);
    k_quant_w<<<(O_N * C_N + 255) / 256, 256, 0, stream>>>(w, wmaxb, wcell);
    k_cim_gemm<<<(MT_TILES * NT_TILES) / 8, 256, 0, stream>>>(xb, wcell, dacc,
                                                              amaxb, wmaxb, y, out);
}
